// Paraformer_43138651521701
// MI455X (gfx1250) — matrix operand order repaired and checked by reading, not compiled, not run on hardware
//
#include <hip/hip_runtime.h>

typedef __attribute__((ext_vector_type(16))) _Float16 v16h;
typedef __attribute__((ext_vector_type(8)))  _Float16 v8h;
typedef __attribute__((ext_vector_type(4)))  _Float16 v4h;
typedef __attribute__((ext_vector_type(8)))  float    v8f;

#define B_   32
#define T_   2048
#define D_   512
#define MAXT 100

// GEMM tiling: M=128 time rows, N=256 out channels, K-chunks of 64 (of 1536)
#define MT  128
#define NT  256
#define KCH 64

// output layout (floats): acoustic_embeds | token_num | alphas | cif_peak
#define AC_OFF 0
#define TN_OFF (B_ * MAXT * D_)            // 1638400
#define AL_OFF (TN_OFF + B_)               // 1638432
#define PK_OFF (AL_OFF + B_ * T_)          // 1703968

// workspace layout (bytes)
#define WSO_BW  0                          // f16 [512][1536] (n-major) = 1572864 B
#define WSO_PL  1572864                    // f32 [2][B][T]   = 524288 B
#define WSO_REM 2097152                    // f32 [B][T]      = 262144 B
#define WSO_ORD 2359296                    // i32 [B][T]      = 262144 B

// ---------------------------------------------------------------------------
// Repack w_conv[o][i][k] (f32) -> Bw2[o][kappa] (f16), kappa = k*512+i.
// n-major so a fixed output channel's K-run is contiguous (b128 staging).
// ---------------------------------------------------------------------------
__global__ __launch_bounds__(256)
void k_pack_w(const float* __restrict__ w_conv, _Float16* __restrict__ Bw2) {
  const int o = blockIdx.x;
#pragma unroll
  for (int j = 0; j < 6; ++j) {
    int kappa = threadIdx.x + j * 256;      // 0..1535
    int k = kappa >> 9;                     // 0..2
    int i = kappa & 511;
    Bw2[(size_t)o * 1536 + kappa] =
        (_Float16)w_conv[((size_t)o * D_ + i) * 3 + k];
  }
}

// ---------------------------------------------------------------------------
// Conv-as-GEMM + bias + ReLU + dot(w_out): WMMA f32<-f16 16x16x32.
// Grid (2, T/128, B), block 256 (8 waves; wave w owns rows w*16..w*16+15
// across the block's full N=256). Writes partial logits -> plog[gx][b][t].
//
// LDS tiles are stored directly in WMMA fragment order (ISA 7.12.2):
//   A frag (16x32 f16): lane = grp*16+m, grp=(kl>>3)&1, half h=(kl>>4)*8+(kl&7)
//   B frag (32x16 f16): lane = grp*16+n, grp=(kl>>3)&1, half h=(kl>>4)*8+(kl&7)
// B tile is staged with GLOBAL_LOAD_ASYNC_TO_LDS_B128 (per-lane swizzled LDS
// dest, ASYNCcnt), overlapped with the A tile's f32->f16 convert+ds_store.
// ---------------------------------------------------------------------------
__global__ __launch_bounds__(256)
void k_conv_gemm(const float* __restrict__ hidden,
                 const _Float16* __restrict__ Bw2,
                 const float* __restrict__ b_conv,
                 const float* __restrict__ w_out,
                 float* __restrict__ plog) {
  __shared__ __attribute__((aligned(32))) _Float16 As[8][2][32][16];   // 16 KB
  __shared__ __attribute__((aligned(32))) _Float16 Bs[16][2][32][16];  // 32 KB

  const int n0  = blockIdx.x * NT;
  const int t0  = blockIdx.y * MT;
  const int b   = blockIdx.z;
  const int tid = threadIdx.x;
  const int lane = tid & 31;
  const int wm  = tid >> 5;            // wave id == M strip 0..7

  v8f acc[16] = {};

  for (int c = 0; c < 24; ++c) {
    const int kconv = c >> 3;            // conv tap 0..2
    const int i0    = (c & 7) * KCH;     // input-channel base

    // ---- stage B: 64x256 f16 fragments via async global->LDS b128 copies
    //      (issued first so the DMA overlaps the A-tile convert below)
#pragma unroll
    for (int r = 0; r < 8; ++r) {
      int u   = tid + r * 256;           // 0..2047 octet units
      int n   = u >> 3;                  // out channel in tile (0..255)
      int kk0 = (u & 7) * 8;             // K octet base (0..56)
      int kb = kk0 >> 5, kl = kk0 & 31;
      int grp = (kl >> 3) & 1;
      unsigned lds_addr =
          (unsigned)(uintptr_t)&Bs[n >> 4][kb][grp * 16 + (n & 15)][(kl >> 4) * 8 + (kl & 7)];
      unsigned goff =
          (unsigned)(((n0 + n) * 1536 + kconv * D_ + i0 + kk0) * 2);
      asm volatile("global_load_async_to_lds_b128 %0, %1, %2"
                   :: "v"(lds_addr), "v"(goff), "s"(Bw2)
                   : "memory");
    }

    // prefetch next chunk's A rows (global_prefetch_b8)
    if (c + 1 < 24) {
      const int nk = (c + 1) >> 3, ni0 = ((c + 1) & 7) * KCH;
      int pm = tid >> 4, ptg = t0 + pm + nk - 1;
      if (ptg >= 0 && ptg < T_)
        __builtin_prefetch(&hidden[((size_t)b * T_ + ptg) * D_ + ni0 + (tid & 15) * 4], 0, 3);
    }

    // ---- stage A: 128x64 f32 -> f16 fragments (float4 loads, b64 stores)
#pragma unroll
    for (int r = 0; r < 8; ++r) {
      int u   = tid + r * 256;           // 0..2047 quad units
      int m   = u >> 4;                  // time row in tile (0..127)
      int kk0 = (u & 15) * 4;            // K quad base (0..60)
      int tg  = t0 + m + kconv - 1;      // conv padding: t-1..t+1
      float4 v = make_float4(0.f, 0.f, 0.f, 0.f);
      if (tg >= 0 && tg < T_)
        v = *(const float4*)&hidden[((size_t)b * T_ + tg) * D_ + i0 + kk0];
      v4h hv;
      hv[0] = (_Float16)v.x; hv[1] = (_Float16)v.y;
      hv[2] = (_Float16)v.z; hv[3] = (_Float16)v.w;
      int kb = kk0 >> 5, kl = kk0 & 31;
      int grp = (kl >> 3) & 1;
      int h0  = (kl >> 4) * 8 + (kl & 7);
      *(v4h*)&As[m >> 4][kb][grp * 16 + (m & 15)][h0] = hv;
    }

    // async B copies must land before the barrier releases readers
    asm volatile("s_wait_asynccnt 0x0" ::: "memory");
    __syncthreads();

    // ---- 32 WMMAs per wave per chunk, 2-deep B-fragment pipeline ----
#pragma unroll
    for (int kb = 0; kb < 2; ++kb) {
      v16h a = *(const v16h*)(&As[wm][kb][lane][0]);
      v16h bf_cur = *(const v16h*)(&Bs[0][kb][lane][0]);
#pragma unroll
      for (int nt = 0; nt < 16; ++nt) {
        v16h bf_nxt;
        if (nt < 15) bf_nxt = *(const v16h*)(&Bs[nt + 1][kb][lane][0]);
        acc[nt] = __builtin_amdgcn_wmma_f32_16x16x32_f16(
            false, a, false, bf_cur, (short)0, acc[nt], false, false);
        bf_cur = bf_nxt;
      }
    }
    __syncthreads();
  }

  // ---- fused epilogue: +b_conv, ReLU, * w_out, reduce over this N-half ----
  const int grp   = lane >> 4;
  const int nlane = lane & 15;
  float s[8] = {};
#pragma unroll
  for (int nt = 0; nt < 16; ++nt) {
    int col = n0 + nt * 16 + nlane;
    float bc = b_conv[col];
    float wo = w_out[col];
#pragma unroll
    for (int r = 0; r < 8; ++r) {
      float v = fmaxf(acc[nt][r] + bc, 0.0f);
      s[r] = fmaf(v, wo, s[r]);
    }
  }
  // deterministic butterfly over the 16 lanes of each half-wave
#pragma unroll
  for (int r = 0; r < 8; ++r) {
    float x = s[r];
    x += __shfl_xor(x, 1, 32);
    x += __shfl_xor(x, 2, 32);
    x += __shfl_xor(x, 4, 32);
    x += __shfl_xor(x, 8, 32);
    s[r] = x;
  }
  if (nlane == 0) {
    // wave wm holds rows t0 + wm*16 + grp*8 + r, fully reduced over N=256
#pragma unroll
    for (int r = 0; r < 8; ++r)
      plog[((size_t)blockIdx.x * B_ + b) * T_ + t0 + wm * 16 + grp * 8 + r] = s[r];
  }
}

// ---------------------------------------------------------------------------
// Per-batch: sigmoid, token_num, alpha scaling, f64 cumsum scan, fire logic.
// Grid B_, block 256 (8 elems/thread).
// ---------------------------------------------------------------------------
__global__ __launch_bounds__(256)
void k_alpha_scan(const float* __restrict__ plog,
                  const float* __restrict__ mask,
                  const int* __restrict__ tll,
                  const float* __restrict__ b_out,
                  float* __restrict__ out_tok,
                  float* __restrict__ out_alpha,
                  float* __restrict__ out_peak,
                  float* __restrict__ ws_rem,
                  int* __restrict__ ws_ord) {
  __shared__ float  red[256];
  __shared__ double dsum[256];
  __shared__ int    isum[256];
  const int b = blockIdx.x;
  const int tid = threadIdx.x;
  const float bo = b_out[0];

  float loc[8];
  float tsum = 0.f;
#pragma unroll
  for (int j = 0; j < 8; ++j) {
    int t = tid * 8 + j;
    float lg = plog[(size_t)b * T_ + t] +
               plog[(size_t)B_ * T_ + (size_t)b * T_ + t] + bo;
    float a = 1.0f / (1.0f + expf(-lg));
    a = fmaxf(a * mask[(size_t)b * T_ + t], 0.0f);
    loc[j] = a;
    tsum += a;
  }
  red[tid] = tsum;
  __syncthreads();
  for (int s = 128; s > 0; s >>= 1) {
    if (tid < s) red[tid] += red[tid + s];
    __syncthreads();
  }
  float token_num = red[0];
  if (tid == 0) out_tok[b] = token_num;
  float scale = (float)tll[b] / token_num;

  double dloc = 0.0;
#pragma unroll
  for (int j = 0; j < 8; ++j) {
    loc[j] *= scale;
    out_alpha[(size_t)b * T_ + tid * 8 + j] = loc[j];
    dloc += (double)loc[j];
  }
  dsum[tid] = dloc;
  __syncthreads();
  for (int off = 1; off < 256; off <<= 1) {   // Hillis-Steele inclusive (f64)
    double add = (tid >= off) ? dsum[tid - off] : 0.0;
    __syncthreads();
    dsum[tid] += add;
    __syncthreads();
  }
  double run = dsum[tid] - dloc;              // exclusive prefix
  float prev_pf = (float)run;
  int fired[8];
#pragma unroll
  for (int j = 0; j < 8; ++j) {
    int t = tid * 8 + j;
    run += (double)loc[j];
    float pf   = (float)run;
    float fl   = floorf(pf);
    float flp  = floorf(prev_pf);             // t==0: run==0 -> 0 (matches ref)
    int fire   = (fl - flp) > 0.0f;
    float frac = pf - fl;
    out_peak[(size_t)b * T_ + t] = (float)fire + frac;
    ws_rem[(size_t)b * T_ + t]   = frac;
    fired[j] = fire;
    prev_pf = pf;
  }
  int ic = fired[0] + fired[1] + fired[2] + fired[3] +
           fired[4] + fired[5] + fired[6] + fired[7];
  isum[tid] = ic;
  __syncthreads();
  for (int off = 1; off < 256; off <<= 1) {   // int scan for ordinals
    int add = (tid >= off) ? isum[tid - off] : 0;
    __syncthreads();
    isum[tid] += add;
    __syncthreads();
  }
  int runi = isum[tid] - ic;
#pragma unroll
  for (int j = 0; j < 8; ++j) {
    int t = tid * 8 + j;
    ws_ord[(size_t)b * T_ + t] = fired[j] ? runi : -1;
    runi += fired[j];
  }
}

// ---------------------------------------------------------------------------
// CIF integrate-and-fire over hidden. Grid B_*4 (D split in 4), block 128.
// out[j] = g_j - g_{j-1}; tail row -g_last; rest zero. Each row written once.
// ---------------------------------------------------------------------------
__global__ __launch_bounds__(128)
void k_cif(const float* __restrict__ hidden,
           const float* __restrict__ alphas,
           const float* __restrict__ ws_rem,
           const int* __restrict__ ws_ord,
           float* __restrict__ out_ac) {
  __shared__ float sa[T_];
  __shared__ float sr[T_];
  __shared__ int   so[T_];
  const int b = blockIdx.x >> 2;
  const int d = (blockIdx.x & 3) * 128 + threadIdx.x;
  for (int i = threadIdx.x; i < T_; i += 128) {
    sa[i] = alphas[(size_t)b * T_ + i];
    sr[i] = ws_rem[(size_t)b * T_ + i];
    so[i] = ws_ord[(size_t)b * T_ + i];
  }
  __syncthreads();

  const float* hp = hidden + (size_t)b * T_ * D_ + d;
  float* op = out_ac + (size_t)b * MAXT * D_ + d;
  float acc = 0.f, prev_g = 0.f;
  int last = -1;
  for (int t = 0; t < T_; ++t) {
    float h = hp[(size_t)t * D_];
    acc = fmaf(sa[t], h, acc);                // cumsum(alpha * hidden)
    int o = so[t];                            // uniform across block
    if (o >= 0) {
      float g = fmaf(-sr[t], h, acc);         // g = prefix - remains*h
      if (o < MAXT) op[(size_t)o * D_] = g - prev_g;
      prev_g = g;
      last = o;
    }
  }
  int jn = last + 1;
  if (jn < MAXT) op[(size_t)jn * D_] = -prev_g;
  for (int j = jn + 1; j < MAXT; ++j) op[(size_t)j * D_] = 0.0f;
}

// ---------------------------------------------------------------------------
extern "C" void kernel_launch(void* const* d_in, const int* in_sizes, int n_in,
                              void* d_out, int out_size, void* d_ws, size_t ws_size,
                              hipStream_t stream) {
  const float* hidden = (const float*)d_in[0];
  const float* mask   = (const float*)d_in[1];
  const int*   tll    = (const int*)d_in[2];
  const float* w_conv = (const float*)d_in[3];
  const float* b_conv = (const float*)d_in[4];
  const float* w_out  = (const float*)d_in[5];
  const float* b_out  = (const float*)d_in[6];
  float* out = (float*)d_out;

  char* ws = (char*)d_ws;
  _Float16* Bw2  = (_Float16*)(ws + WSO_BW);
  float*    plog = (float*)(ws + WSO_PL);
  float*    remw = (float*)(ws + WSO_REM);
  int*      ordw = (int*)(ws + WSO_ORD);

  // 1) repack conv weights to f16 GEMM-B (n-major)
  k_pack_w<<<D_, 256, 0, stream>>>(w_conv, Bw2);

  // 2) conv-as-GEMM via WMMA + fused logit projection
  dim3 gA(D_ / NT, T_ / MT, B_);              // (2, 16, 32)
  k_conv_gemm<<<gA, 256, 0, stream>>>(hidden, Bw2, b_conv, w_out, plog);

  // 3) per-batch sigmoid / normalize / f64 scan / fire detection
  k_alpha_scan<<<B_, 256, 0, stream>>>(plog, mask, tll, b_out,
                                       out + TN_OFF, out + AL_OFF, out + PK_OFF,
                                       remw, ordw);

  // 4) CIF integration + scatter into token slots
  k_cif<<<B_ * 4, 128, 0, stream>>>(hidden, out + AL_OFF, remw, ordw,
                                    out + AC_OFF);
}
